// GraphDecoder_86268713108206
// MI455X (gfx1250) — compile-verified, matrix operand order checked by reading, not run on hardware
//
#include <hip/hip_runtime.h>

// ---------------------------------------------------------------------------
// Types for CDNA5 WMMA (wave32): v16bf A/B fragments, v8f accumulator.
// ---------------------------------------------------------------------------
typedef __attribute__((ext_vector_type(16))) __bf16 v16bf;
typedef __attribute__((ext_vector_type(8)))  float  v8f;
typedef __attribute__((ext_vector_type(4)))  unsigned int v4u;
typedef __attribute__((ext_vector_type(8)))  unsigned int v8u;

__device__ __forceinline__ unsigned short f2bf(float f) {
  union { float f; unsigned int u; } v; v.f = f;
  unsigned int r = v.u + 0x7FFFu + ((v.u >> 16) & 1u);   // round-to-nearest-even
  return (unsigned short)(r >> 16);
}

// ---------------------------------------------------------------------------
// Activation buffer layout = WMMA A-fragment panel order:
//   panel p = node/16 (16 rows), per panel: [ktile][lane][16 bf16]
//   element (row m, K): khalf=(K>>3)&1, lane=m+16*khalf, t=(K&7)+8*((K>>4)&1)
// This makes each block's A panel one contiguous 16*Kpad*2-byte chunk ->
// a single 1-D TDM tensor_load_to_lds per block.
// ---------------------------------------------------------------------------
__device__ __forceinline__ size_t afrag_idx(long long node, int K, int Kpad) {
  long long p = node >> 4;
  int m  = (int)node & 15;
  int kt = K >> 5, kl = K & 31;
  int lane = m + (((kl >> 3) & 1) << 4);
  int t    = (kl & 7) + (((kl >> 4) & 1) << 3);
  return ((size_t)p * (Kpad >> 5) + kt) * 512 + (size_t)lane * 16 + t;
}

// ---------------------------------------------------------------------------
// Pack f32 weight matrix [Kreal x Cout] (row-major) into WMMA B-fragment
// order: P[(kt*ntiles + nt)*512 + lane*16 + t] (mirrors A layout, indexed by
// output column instead of row). Rows k >= Kreal are zero padding.
// ---------------------------------------------------------------------------
__global__ void k_pack_w(const float* __restrict__ W, unsigned short* __restrict__ P,
                         int Kreal, int Kpad, int Cout) {
  int ntiles = Cout >> 4;
  int total  = (Kpad >> 5) * ntiles * 512;
  int i = blockIdx.x * blockDim.x + threadIdx.x;
  if (i >= total) return;
  int t     = i & 15;
  int lane  = (i >> 4) & 31;
  int rest  = i >> 9;
  int nt    = rest % ntiles;
  int kt    = rest / ntiles;
  int khalf = lane >> 4;
  int k = kt * 32 + khalf * 8 + (t & 7) + ((t >= 8) ? 16 : 0);
  int c = nt * 16 + (lane & 15);
  float w = (k < Kreal) ? W[(size_t)k * Cout + c] : 0.0f;
  P[i] = f2bf(w);
}

// ---------------------------------------------------------------------------
// Y[N x Cout] = A_bf16 @ W_packed + bias.
// Per block: TDM DMAs the contiguous 16-row A panel into LDS (one
// tensor_load_to_lds issued by wave 0, waited with s_wait_tensorcnt, released
// with the workgroup barrier). 8 waves then each own 16x16 output tiles; the
// K-loop is two ds_load_b128 (A) + one 32B global load (B, L2-resident,
// prefetched) per v_wmma_f32_16x16x32_bf16.
// ---------------------------------------------------------------------------
__global__ void __launch_bounds__(256)
k_wmma_gemm(const unsigned short* __restrict__ A,   // panel-fragment order
            const unsigned short* __restrict__ Bp,  // [nkt][ntiles][32][16]
            const float* __restrict__ bias, float* __restrict__ Y,
            int N, int Kpad, int Cout) {
  __shared__ unsigned short shA[16 * 352];          // up to 11264 B
  const int lane   = threadIdx.x & 31;
  const int wave   = threadIdx.x >> 5;
  const int row0   = blockIdx.x << 4;
  const int ntiles = Cout >> 4;
  const int nkt    = Kpad >> 5;

  if (wave == 0) {
    // ---- Tensor DMA descriptor (ISA 8.3/8.4), all wave-uniform -> SGPRs ----
    const unsigned long long gaddr =
        (unsigned long long)(const unsigned short*)(A + (size_t)blockIdx.x * 16 * Kpad);
    const unsigned ldsb = (unsigned)(unsigned long long)(void*)&shA[0];
    const unsigned td0  = (unsigned)(Kpad << 3);    // 16*Kpad bf16 = 8*Kpad dwords
    const v4u g0 = { 1u,                            // count=1, user descriptor
                     ldsb,                          // lds_addr (bytes)
                     (unsigned)gaddr,               // global_addr[31:0]
                     (unsigned)((gaddr >> 32) & 0x01FFFFFFu) | (2u << 30) }; // type=2
    const v8u g1 = { (2u << 16),                    // data_size = 4B, no pad/iter
                     (td0 & 0xFFFFu) << 16,         // tensor_dim0 lo16
                     (td0 >> 16) | (1u << 16),      // tensor_dim0 hi | tensor_dim1=1
                     (td0 & 0xFFFFu) << 16,         // tile_dim0 = whole panel (1-D)
                     0u,                            // tile_dim1=0, tile_dim2=0
                     td0,                           // tensor_dim0_stride lo
                     0u, 0u };
    const v4u gz = { 0u, 0u, 0u, 0u };              // groups 2/3: dims 2+ unused
    asm volatile("tensor_load_to_lds %0, %1, %2, %3"
                 :: "s"(g0), "s"(g1), "s"(gz), "s"(gz)
                 : "memory");
    __builtin_amdgcn_s_wait_tensorcnt(0);
  }
  __syncthreads();

  for (int nt = wave; nt < ntiles; nt += 8) {
    v8f acc = {};
    const unsigned short* aptr = shA + lane * 16;
    const unsigned short* bptr = Bp + ((size_t)nt * 32 + lane) * 16;
    for (int kt = 0; kt < nkt; ++kt) {
      if (kt + 1 < nkt)   // keep the B stream ahead (global_prefetch_b8)
        __builtin_prefetch((const void*)(bptr + (size_t)ntiles * 512), 0, 1);
      v16bf a = *(const v16bf*)aptr;                // 2x ds_load_b128
      v16bf b = *(const v16bf*)bptr;                // 32B global, L2-hot
      aptr += 512;
      bptr += (size_t)ntiles * 512;
      acc = __builtin_amdgcn_wmma_f32_16x16x32_bf16(false, a, false, b,
                                                    (short)0, acc, false, false);
    }
    const int m = lane & 15;
    const int col = (nt << 4) + m;
    const float bv = bias[col];
    const int rbase = (lane >> 4) << 3;
    float* yp = Y + (size_t)row0 * Cout + col;
#pragma unroll
    for (int r = 0; r < 8; ++r) {
      int row = row0 + rbase + r;
      if (row < N) yp[(size_t)(rbase + r) * Cout] = acc[r] + bv;
    }
  }
}

// ---------------------------------------------------------------------------
// Edge aggregation: y[a] += n[b], y[b] += n[a] for every undirected edge,
// replicated over nmesh meshes (vertex offset mesh*V). float4-vectorized
// channels, scalar f32 global atomics (avg degree ~6 -> low contention).
// ---------------------------------------------------------------------------
__global__ void k_edge_scatter4(const int* __restrict__ edges, int E, int V, int nmesh,
                                const float* __restrict__ nf, float* __restrict__ y, int C) {
  long long tid = (long long)blockIdx.x * blockDim.x + threadIdx.x;
  int c4 = C >> 2;
  long long total = (long long)E * nmesh * c4;
  if (tid >= total) return;
  int cpart = (int)(tid % c4);
  long long e = tid / c4;
  int mesh = (int)(e / E);
  int ee   = (int)(e - (long long)mesh * E);
  int a = edges[2 * ee]     + mesh * V;
  int b = edges[2 * ee + 1] + mesh * V;
  int c0 = cpart << 2;
  const float4 na = *(const float4*)(nf + (size_t)a * C + c0);
  const float4 nb = *(const float4*)(nf + (size_t)b * C + c0);
  float* ya = y + (size_t)a * C + c0;
  float* yb = y + (size_t)b * C + c0;
  atomicAdd(ya + 0, nb.x); atomicAdd(ya + 1, nb.y);
  atomicAdd(ya + 2, nb.z); atomicAdd(ya + 3, nb.w);
  atomicAdd(yb + 0, na.x); atomicAdd(yb + 1, na.y);
  atomicAdd(yb + 2, na.z); atomicAdd(yb + 3, na.w);
}

__global__ void k_edge_scatter_s(const int* __restrict__ edges, int E, int V, int nmesh,
                                 const float* __restrict__ nf, float* __restrict__ y, int C) {
  long long tid = (long long)blockIdx.x * blockDim.x + threadIdx.x;
  long long total = (long long)E * nmesh * C;
  if (tid >= total) return;
  int c = (int)(tid % C);
  long long e = tid / C;
  int mesh = (int)(e / E);
  int ee   = (int)(e - (long long)mesh * E);
  int a = edges[2 * ee]     + mesh * V;
  int b = edges[2 * ee + 1] + mesh * V;
  atomicAdd(&y[(size_t)a * C + c], nf[(size_t)b * C + c]);
  atomicAdd(&y[(size_t)b * C + c], nf[(size_t)a * C + c]);
}

// ---------------------------------------------------------------------------
// Post-GC pointwise: v = y (after scatter); optional ReLU; optional +residual;
// write f32 copy and/or bf16 copy in A-fragment panel order (column offset
// col0, padded K = Kpad) feeding the next WMMA GEMM.
// ---------------------------------------------------------------------------
__global__ void k_postproc(const float* __restrict__ y, const float* __restrict__ resid,
                           float* __restrict__ of32, unsigned short* __restrict__ obf,
                           long long Ntot, int C, int Kpad, int col0, int relu) {
  long long i = (long long)blockIdx.x * blockDim.x + threadIdx.x;
  long long total = Ntot * C;
  if (i >= total) return;
  int c = (int)(i % C);
  long long n = i / C;
  float v = y[i];
  if (relu) v = fmaxf(v, 0.0f);
  if (resid) v += resid[i];
  if (of32) of32[i] = v;
  if (obf) obf[afrag_idx(n, col0 + c, Kpad)] = f2bf(v);
}

// ---------------------------------------------------------------------------
// Trilinear grid_sample (align_corners=True) of skip volume [B,Cs,D,H,W] at
// coords = verts; writes bf16 directly into the concat feature panels.
// ---------------------------------------------------------------------------
__global__ void k_trilinear(const float* __restrict__ vol, const float* __restrict__ verts,
                            unsigned short* __restrict__ obf, int MV, int Cs,
                            int D, int H, int W, int Kpad, int col0, long long Ntot) {
  long long i = (long long)blockIdx.x * blockDim.x + threadIdx.x;
  long long total = Ntot * Cs;
  if (i >= total) return;
  int c = (int)(i % Cs);
  long long node = i / Cs;
  int b = (int)(node / MV);
  float cx = verts[node * 3 + 0];
  float cy = verts[node * 3 + 1];
  float cz = verts[node * 3 + 2];
  float fx = (cx + 1.0f) * 0.5f * (float)(W - 1);
  float fy = (cy + 1.0f) * 0.5f * (float)(H - 1);
  float fz = (cz + 1.0f) * 0.5f * (float)(D - 1);
  int x0 = (int)floorf(fx); x0 = x0 < 0 ? 0 : (x0 > W - 2 ? W - 2 : x0);
  int y0 = (int)floorf(fy); y0 = y0 < 0 ? 0 : (y0 > H - 2 ? H - 2 : y0);
  int z0 = (int)floorf(fz); z0 = z0 < 0 ? 0 : (z0 > D - 2 ? D - 2 : z0);
  float tx = fx - (float)x0, ty = fy - (float)y0, tz = fz - (float)z0;
  const float* vb = vol + ((size_t)b * Cs + c) * ((size_t)D * H * W);
  size_t s00 = ((size_t)z0 * H + y0) * W + x0;
  size_t s10 = s00 + (size_t)H * W;
  float v000 = vb[s00],     v001 = vb[s00 + 1];
  float v010 = vb[s00 + W], v011 = vb[s00 + W + 1];
  float v100 = vb[s10],     v101 = vb[s10 + 1];
  float v110 = vb[s10 + W], v111 = vb[s10 + W + 1];
  float v = v000 * (1 - tz) * (1 - ty) * (1 - tx) + v001 * (1 - tz) * (1 - ty) * tx
          + v010 * (1 - tz) * ty * (1 - tx)       + v011 * (1 - tz) * ty * tx
          + v100 * tz * (1 - ty) * (1 - tx)       + v101 * tz * (1 - ty) * tx
          + v110 * tz * ty * (1 - tx)             + v111 * tz * ty * tx;
  obf[afrag_idx(node, col0 + c, Kpad)] = f2bf(v);
}

// Write verts (3 cols at col0) + zero the pad columns [zero_from, Kpad).
__global__ void k_emit_tail(const float* __restrict__ verts, unsigned short* __restrict__ obf,
                            long long Ntot, int Kpad, int col0, int zero_from) {
  int width = 3 + (Kpad - zero_from);
  long long i = (long long)blockIdx.x * blockDim.x + threadIdx.x;
  long long total = Ntot * width;
  if (i >= total) return;
  int j = (int)(i % width);
  long long node = i / width;
  if (j < 3) obf[afrag_idx(node, col0 + j, Kpad)] = f2bf(verts[node * 3 + j]);
  else       obf[afrag_idx(node, zero_from + (j - 3), Kpad)] = 0;
}

__global__ void k_init_verts(const float* __restrict__ tmpl, float* __restrict__ verts,
                             int MV3, long long Ntot3) {
  long long i = (long long)blockIdx.x * blockDim.x + threadIdx.x;
  if (i >= Ntot3) return;
  verts[i] = tmpl[i % MV3];
}

// f2v GraphConv (Cout=3) — tiny, scalar.  yv = x@w0+b0, nv = x@w1+b1.
__global__ void k_f2v(const float* __restrict__ x,
                      const float* __restrict__ w0, const float* __restrict__ b0,
                      const float* __restrict__ w1, const float* __restrict__ b1,
                      float* __restrict__ yv, float* __restrict__ nv,
                      long long Ntot, int Cin) {
  long long i = (long long)blockIdx.x * blockDim.x + threadIdx.x;
  long long total = Ntot * 3;
  if (i >= total) return;
  int oc = (int)(i % 3);
  long long node = i / 3;
  const float* xr = x + node * Cin;
  float s0 = b0[oc], s1 = b1[oc];
  for (int k = 0; k < Cin; ++k) {
    float xv = xr[k];
    s0 += xv * w0[k * 3 + oc];
    s1 += xv * w1[k * 3 + oc];
  }
  yv[i] = s0; nv[i] = s1;
}

__global__ void k_update_verts(float* __restrict__ verts, float* __restrict__ dV,
                               const float* __restrict__ yv, long long n3) {
  long long i = (long long)blockIdx.x * blockDim.x + threadIdx.x;
  if (i >= n3) return;
  float d = yv[i];
  dV[i] = d;
  verts[i] += d;
}

// out[node, 0:128]=latent, [128:131]=verts, [131:134]=deltaV
__global__ void k_assemble(const float* __restrict__ latent, const float* __restrict__ verts,
                           const float* __restrict__ dV, float* __restrict__ out,
                           long long Ntot, int Clat) {
  int Ctot = Clat + 6;
  long long i = (long long)blockIdx.x * blockDim.x + threadIdx.x;
  long long total = Ntot * Ctot;
  if (i >= total) return;
  int j = (int)(i % Ctot);
  long long node = i / Ctot;
  float v;
  if (j < Clat)          v = latent[node * Clat + j];
  else if (j < Clat + 3) v = verts[node * 3 + (j - Clat)];
  else                   v = dV[node * 3 + (j - Clat - 3)];
  out[i] = v;
}

// ---------------------------------------------------------------------------
// Host orchestration
// ---------------------------------------------------------------------------
extern "C" void kernel_launch(void* const* d_in, const int* in_sizes, int n_in,
                              void* d_out, int out_size, void* d_ws, size_t ws_size,
                              hipStream_t stream) {
  (void)out_size; (void)ws_size;
  if (n_in < 137) return;

  const int Bb = 2, Mm = 4, Vv = 40002;
  const long long Ntot = (long long)Bb * Mm * Vv;   // 320016 (divisible by 16)
  const int nmesh = Bb * Mm;
  const int E = in_sizes[136] / 2;                  // 120000

  // ---- parameter leaves: pytree flatten with sorted dict keys -------------
  auto P = [&](int i) { return (const float*)d_in[i]; };
  struct GCp { const float *b0, *b1, *w0, *w1; };
  int pi = 0;
  GCp first[3];
  for (int i = 0; i < 3; ++i) { first[i] = { P(pi), P(pi+1), P(pi+2), P(pi+3) }; pi += 4; }
  GCp blocks[3][3][3]; GCp f2v[3];
  for (int s = 0; s < 3; ++s) {
    for (int b = 0; b < 3; ++b)
      for (int g = 0; g < 3; ++g) { blocks[s][b][g] = { P(pi), P(pi+1), P(pi+2), P(pi+3) }; pi += 4; }
    f2v[s] = { P(pi), P(pi+1), P(pi+2), P(pi+3) }; pi += 4;
  }
  const float* skips[3] = { P(132), P(133), P(134) };
  const float* vtmpl = P(135);
  const int* edges = (const int*)d_in[136];

  // ---- workspace carve (deterministic) ------------------------------------
  char* base = (char*)d_ws;
  size_t off = 0;
  auto alloc = [&](size_t bytes) -> void* {
    void* p = base + off;
    off += (bytes + 255) & ~(size_t)255;
    return p;
  };
  const int KPAD_MAX = 352;                         // pad32(256+64+3)
  unsigned short* abf  = (unsigned short*)alloc((size_t)Ntot * KPAD_MAX * 2);
  float* ybuf = (float*)alloc((size_t)Ntot * 256 * 4);
  float* nbuf = (float*)alloc((size_t)Ntot * 256 * 4);
  float* h0f  = (float*)alloc((size_t)Ntot * 256 * 4);
  float* curf = (float*)alloc((size_t)Ntot * 256 * 4);
  float* verts = (float*)alloc((size_t)Ntot * 3 * 4);
  float* dV    = (float*)alloc((size_t)Ntot * 3 * 4);
  float* yv    = (float*)alloc((size_t)Ntot * 3 * 4);
  float* nv    = (float*)alloc((size_t)Ntot * 3 * 4);
  unsigned short* pw0 = (unsigned short*)alloc((size_t)KPAD_MAX * 256 * 2);
  unsigned short* pw1 = (unsigned short*)alloc((size_t)KPAD_MAX * 256 * 2);

  auto cdiv = [](long long a, long long b) { return (unsigned)((a + b - 1) / b); };
  const unsigned gridM = (unsigned)(Ntot / 16);     // 20001 row panels

  // One GraphConv: pack weights -> 2 WMMA GEMMs -> edge scatter -> postproc.
  auto run_gc = [&](const GCp& g, int Kreal, int Kpad, int Cout, int relu,
                    const float* resid, float* of32,
                    unsigned short* obf, int obf_kpad, int obf_col0) {
    long long pt = (long long)(Kpad / 32) * (Cout / 16) * 512;
    k_pack_w<<<cdiv(pt, 256), 256, 0, stream>>>(g.w0, pw0, Kreal, Kpad, Cout);
    k_pack_w<<<cdiv(pt, 256), 256, 0, stream>>>(g.w1, pw1, Kreal, Kpad, Cout);
    k_wmma_gemm<<<gridM, 256, 0, stream>>>(abf, pw0, g.b0, ybuf, (int)Ntot, Kpad, Cout);
    k_wmma_gemm<<<gridM, 256, 0, stream>>>(abf, pw1, g.b1, nbuf, (int)Ntot, Kpad, Cout);
    long long st = (long long)E * nmesh * (Cout / 4);
    k_edge_scatter4<<<cdiv(st, 256), 256, 0, stream>>>(edges, E, Vv, nmesh, nbuf, ybuf, Cout);
    long long pp = Ntot * Cout;
    k_postproc<<<cdiv(pp, 256), 256, 0, stream>>>(ybuf, resid, of32, obf, Ntot, Cout,
                                                  obf_kpad, obf_col0, relu);
  };

  // ---- verts init + first f2f_res (3 -> 256 -> 256 -> 256, residual) ------
  k_init_verts<<<cdiv(Ntot * 3, 256), 256, 0, stream>>>(vtmpl, verts, Mm * Vv * 3, Ntot * 3);
  k_emit_tail<<<cdiv(Ntot * 32, 256), 256, 0, stream>>>(verts, abf, Ntot, 32, 0, 3);

  const int SKIPC[3] = { 64, 32, 16 };
  const int SDIM[3]  = { 32, 64, 128 };
  const int Kcat0 = 256 + 64 + 3;                   // 323 -> pad 352
  run_gc(first[0],   3,  32, 256, 1, nullptr, h0f,  abf, 256, 0);
  run_gc(first[1], 256, 256, 256, 1, nullptr, curf, abf, 256, 0);
  run_gc(first[2], 256, 256, 256, 1, h0f,     curf, abf, (Kcat0 + 31) / 32 * 32, 0);

  // ---- three refinement steps ---------------------------------------------
  int Clat = 256;
  for (int s = 0; s < 3; ++s) {
    const int Cs = SKIPC[s], D = SDIM[s];
    const int Kcat = Clat + Cs + 3;
    const int Kpad = (Kcat + 31) / 32 * 32;
    // latent bf16 already occupies cols [0,Clat) in Kpad-panels (prev postproc)
    k_trilinear<<<cdiv(Ntot * Cs, 256), 256, 0, stream>>>(skips[s], verts, abf, Mm * Vv, Cs,
                                                          D, D, D, Kpad, Clat, Ntot);
    k_emit_tail<<<cdiv(Ntot * (3 + Kpad - Kcat), 256), 256, 0, stream>>>(
        verts, abf, Ntot, Kpad, Clat + Cs, Kcat);

    for (int b = 0; b < 3; ++b) {
      const int kin    = (b == 0) ? Kcat : 128;
      const int kpadin = (b == 0) ? Kpad : 128;
      run_gc(blocks[s][b][0], kin, kpadin, 128, 1, nullptr, h0f,  abf, 128, 0);
      run_gc(blocks[s][b][1], 128, 128,    128, 1, nullptr, curf, abf, 128, 0);
      const bool stepLast = (b == 2);
      int nextKpad = 128;
      unsigned short* nextBf = abf;
      if (stepLast) {
        if (s < 2) nextKpad = (128 + SKIPC[s + 1] + 3 + 31) / 32 * 32;
        else       nextBf = nullptr;                 // latent not consumed as bf16 again
      }
      run_gc(blocks[s][b][2], 128, 128, 128, 1, h0f, curf, nextBf, nextKpad, 0);
    }

    // f2v (128 -> 3, zero-init weights) + edge scatter + verts update
    k_f2v<<<cdiv(Ntot * 3, 256), 256, 0, stream>>>(curf, f2v[s].w0, f2v[s].b0,
                                                   f2v[s].w1, f2v[s].b1, yv, nv, Ntot, 128);
    long long st3 = (long long)E * nmesh * 3;
    k_edge_scatter_s<<<cdiv(st3, 256), 256, 0, stream>>>(edges, E, Vv, nmesh, nv, yv, 3);
    k_update_verts<<<cdiv(Ntot * 3, 256), 256, 0, stream>>>(verts, dV, yv, Ntot * 3);
    Clat = 128;
  }

  // ---- final output: [latent(128) | verts(3) | deltaV(3)] -----------------
  k_assemble<<<cdiv(Ntot * 134, 256), 256, 0, stream>>>(curf, verts, dV,
                                                        (float*)d_out, Ntot, 128);
}
